// RegionCodecDict_9028021256393
// MI455X (gfx1250) — matrix operand order repaired and checked by reading, not compiled, not run on hardware
//
#include <hip/hip_runtime.h>
#include <hip/hip_bf16.h>

// ---------------------------------------------------------------------------
// RegionCodec fused encode->decode with bf16 WMMA on gfx1250 (MI455X).
// B=8, T=1024 (BT=8192), N=1024, R=8, NR=128, D=512.
// ---------------------------------------------------------------------------

typedef __attribute__((ext_vector_type(16))) __bf16 v16bf;
typedef __attribute__((ext_vector_type(8)))  __bf16 v8bf;
typedef __attribute__((ext_vector_type(8)))  float  v8f;

#define WMMA_BF16(A, B, C) \
    __builtin_amdgcn_wmma_f32_16x16x32_bf16(false, (A), false, (B), (short)0, (C), false, false)

// ---------------------------------------------------------------------------
// Weight prep: f32 (K-major) -> bf16 fragment-swizzled layout.
// B-fragment lane layout (16-bit, 32x16 K x N tile):
//   lane L holds column n = L%16, K = kt*32 + (L/16)*16 + e, e = 0..15
// Stored so each lane's 16 bf16 elements are contiguous (one 32B load).
// ---------------------------------------------------------------------------

// W_enc: per region (NR=128) x (D=512), K=NR.  nt: 0..31, kt: 0..3.
__global__ __launch_bounds__(256) void prep_enc_kernel(const float* __restrict__ W,
                                                       __bf16* __restrict__ outw) {
    int tid = blockIdx.x * 256 + threadIdx.x;      // 524288 total
    int e    = tid & 15;
    int lane = (tid >> 4) & 31;
    int kt   = (tid >> 9) & 3;
    int nt   = (tid >> 11) & 31;
    int r    = tid >> 16;
    int n = nt * 16 + (lane & 15);
    int k = kt * 32 + (lane >> 4) * 16 + e;
    outw[tid] = (__bf16)W[(r * 128 + k) * 512 + n];
}

// W_dec: per region (D=512) x (NR=128), K=D.  nt: 0..7, kt: 0..15.
__global__ __launch_bounds__(256) void prep_dec_kernel(const float* __restrict__ W,
                                                       __bf16* __restrict__ outw) {
    int tid = blockIdx.x * 256 + threadIdx.x;      // 524288 total
    int e    = tid & 15;
    int lane = (tid >> 4) & 31;
    int kt   = (tid >> 9) & 15;
    int nt   = (tid >> 13) & 7;
    int r    = tid >> 16;
    int n = nt * 16 + (lane & 15);
    int k = kt * 32 + (lane >> 4) * 16 + e;
    outw[tid] = (__bf16)W[(r * 512 + k) * 128 + n];
}

// ---------------------------------------------------------------------------
// A-fragment loader (16-bit A, 16x32): lane holds row m = lane%16;
// K elements: contiguous run of 16 at (g*16), g = lane/16.
// rowbase points to element (row, k0) of a row-major bf16 matrix in LDS.
// ---------------------------------------------------------------------------
__device__ __forceinline__ v16bf load_a_frag(const __bf16* rowbase, int g) {
    v8bf lo = *(const v8bf*)(rowbase + g * 16);
    v8bf hi = *(const v8bf*)(rowbase + g * 16 + 8);
    return __builtin_shufflevector(lo, hi, 0, 1, 2, 3, 4, 5, 6, 7,
                                   8, 9, 10, 11, 12, 13, 14, 15);
}

// ---------------------------------------------------------------------------
// Fused main kernel. grid.x = (BT/64) * R = 128 * 8 = 1024, block = 256.
// ---------------------------------------------------------------------------
__global__ __launch_bounds__(256) void region_codec_kernel(
    const float* __restrict__ spikes,     // (8192, 1024)
    const float* __restrict__ b_enc,      // (8, 512)
    const float* __restrict__ b_dec,      // (8, 128)
    const int*   __restrict__ ridx,       // (8, 128)
    const __bf16* __restrict__ WencF,     // prepped, 8*128*512
    const __bf16* __restrict__ WdecF,     // prepped, 8*512*128
    float* __restrict__ out)              // (8192, 1024)
{
    // Padded strides (+8 bf16 = +4 dwords) -> 4-bank rotation per row, so the
    // 16 distinct rows of an A-fragment b128 load hit all 64 banks.
    __shared__ __align__(16) __bf16 sX[64][136];    // gathered X tile (bf16)
    __shared__ __align__(16) __bf16 sTok[64][520];  // tokens tile (bf16)
    __shared__ int sIdx[128];

    const int t    = threadIdx.x;
    const int lane = t & 31;
    const int wave = t >> 5;
    const int band = wave & 3;    // 16-row band: rows [band*16, band*16+16)
    const int half = wave >> 2;   // N-half of D in stage 1 / of NR in stage 2
    const int g    = lane >> 4;
    const int n16  = lane & 15;

    const int r  = blockIdx.x & 7;
    const int m0 = (blockIdx.x >> 3) * 64;

    if (t < 128) sIdx[t] = ridx[r * 128 + t];
    __syncthreads();

    // ---- gather X tile: 64 rows x 128 channels, f32 -> bf16 ----
    for (int e = t; e < 64 * 128; e += 256) {
        int row = e >> 7;
        int c   = e & 127;
        sX[row][c] = (__bf16)spikes[(size_t)(m0 + row) * 1024 + sIdx[c]];
    }
    __syncthreads();

    // =========================== Stage 1: encode ===========================
    // tokens[16 rows of band][512] ; this wave covers d in [half*256, +256).
    {
        const __bf16* xrow = &sX[band * 16 + n16][0];
        v16bf aF[4];
#pragma unroll
        for (int kt = 0; kt < 4; ++kt)
            aF[kt] = load_a_frag(xrow + kt * 32, g);

        const __bf16* wbase = WencF + r * 65536;
        const v8f zero = {0.f, 0.f, 0.f, 0.f, 0.f, 0.f, 0.f, 0.f};

#pragma unroll
        for (int blk = 0; blk < 4; ++blk) {
            v8f acc[4];
#pragma unroll
            for (int j = 0; j < 4; ++j) acc[j] = zero;
#pragma unroll
            for (int kt = 0; kt < 4; ++kt) {
#pragma unroll
                for (int j = 0; j < 4; ++j) {
                    int nt = half * 16 + blk * 4 + j;
                    v16bf b = *(const v16bf*)(wbase + (nt * 4 + kt) * 512 + lane * 16);
                    acc[j] = WMMA_BF16(aF[kt], b, acc[j]);
                }
            }
#pragma unroll
            for (int j = 0; j < 4; ++j) {
                int nt = half * 16 + blk * 4 + j;
                float bias = b_enc[r * 512 + nt * 16 + n16];
#pragma unroll
                for (int v = 0; v < 8; ++v) {
                    // C layout: row m = v + 8*g, col = n16
                    sTok[band * 16 + v + 8 * g][nt * 16 + n16] =
                        (__bf16)(acc[j][v] + bias);
                }
            }
        }
    }
    __syncthreads();

    // =========================== Stage 2: decode ===========================
    // y[16 rows of band][128] ; this wave covers n in [half*64, +64).
    {
        const __bf16* trow = &sTok[band * 16 + n16][0];
        const __bf16* wdbase = WdecF + r * 65536;
        const v8f zero = {0.f, 0.f, 0.f, 0.f, 0.f, 0.f, 0.f, 0.f};
        v8f acc[4];
#pragma unroll
        for (int j = 0; j < 4; ++j) acc[j] = zero;

#pragma unroll
        for (int kt = 0; kt < 16; ++kt) {
            v16bf a = load_a_frag(trow + kt * 32, g);
#pragma unroll
            for (int j = 0; j < 4; ++j) {
                int nt = half * 4 + j;
                v16bf b = *(const v16bf*)(wdbase + (nt * 16 + kt) * 512 + lane * 16);
                acc[j] = WMMA_BF16(a, b, acc[j]);
            }
        }

#pragma unroll
        for (int j = 0; j < 4; ++j) {
            int nt = half * 4 + j;
            int c  = nt * 16 + n16;
            float bias = b_dec[r * 128 + c];
            int col = sIdx[c];
            int rowg = m0 + band * 16 + 8 * g;
#pragma unroll
            for (int v = 0; v < 8; ++v) {
                out[(size_t)(rowg + v) * 1024 + col] = acc[j][v] + bias;
            }
        }
    }
}

// ---------------------------------------------------------------------------
extern "C" void kernel_launch(void* const* d_in, const int* in_sizes, int n_in,
                              void* d_out, int out_size, void* d_ws, size_t ws_size,
                              hipStream_t stream) {
    const float* spikes = (const float*)d_in[0];   // (8,1024,1024)
    const float* W_enc  = (const float*)d_in[1];   // (8,128,512)
    const float* b_enc  = (const float*)d_in[2];   // (8,512)
    const float* W_dec  = (const float*)d_in[3];   // (8,512,128)
    const float* b_dec  = (const float*)d_in[4];   // (8,128)
    const int*   ridx   = (const int*)d_in[5];     // (8,128)
    float* out = (float*)d_out;

    __bf16* WencF = (__bf16*)d_ws;                 // 1 MB
    __bf16* WdecF = WencF + 8 * 128 * 512;         // 1 MB  (needs ws >= 2 MB)

    prep_enc_kernel<<<2048, 256, 0, stream>>>(W_enc, WencF);
    prep_dec_kernel<<<2048, 256, 0, stream>>>(W_dec, WdecF);

    // (BT/64) * R = 128 * 8 = 1024 workgroups, 8 waves each.
    region_codec_kernel<<<1024, 256, 0, stream>>>(spikes, b_enc, b_dec, ridx,
                                                  WencF, WdecF, out);
}
